// TensoRF_43138651521250
// MI455X (gfx1250) — matrix operand order repaired and checked by reading, not compiled, not run on hardware
//
#include <hip/hip_runtime.h>
#include <hip/hip_fp16.h>
#include <math.h>

typedef __attribute__((ext_vector_type(16))) _Float16 v16h;
typedef __attribute__((ext_vector_type(8)))  _Float16 v8h;
typedef __attribute__((ext_vector_type(8)))  float    v8f;

#define G 128
#define NRAYS 8192
#define NSAMP 64
#define TOTAL (NRAYS*NSAMP)

#define WAVES 4
#define BLOCK (WAVES*32)
#define SPB   (WAVES*16)      // samples per block

// LDS row strides in halves; multiples of 8 so every 8-half group is 16B aligned,
// and not multiples of 128 so consecutive rows rotate across the 64 LDS banks.
#define H1S 144               // h1 [16][H1S]  (288B rows)
#define CNS 80                // cin / h2 [16][CNS] (160B rows)

static __device__ __forceinline__ v8f wmma_f16(v16h a, v16h b, v8f c) {
  return __builtin_amdgcn_wmma_f32_16x16x32_f16(false, a, false, b, (short)0, c, false, false);
}
static __device__ __forceinline__ v16h cat8(v8h lo, v8h hi) {
  return __builtin_shufflevector(lo, hi, 0,1,2,3,4,5,6,7,8,9,10,11,12,13,14,15);
}

__global__ __launch_bounds__(BLOCK) void nerf_point_kernel(
    const float* __restrict__ rays_o, const float* __restrict__ rays_d,
    const float* __restrict__ aabb,   const float* __restrict__ dgrid,
    const float* __restrict__ cfeat,
    const float* __restrict__ dW1, const float* __restrict__ db1,
    const float* __restrict__ dW2, const float* __restrict__ db2,
    const float* __restrict__ cW1, const float* __restrict__ cb1,
    const float* __restrict__ cW2, const float* __restrict__ cb2,
    const float* __restrict__ cW3, const float* __restrict__ cb3,
    float* __restrict__ ws_sigma, float* __restrict__ ws_rgb)
{
  // Weights pre-swizzled into WMMA B-fragment order:
  // sW1p[((kc*8 + nt)*32 + lane)*16 + i] = cW1[K = kc*32 + (lane>>4)*16 + i][N = nt*16 + (lane&15)]
  __shared__ __align__(32) _Float16 sW1p[2 * 8 * 32 * 16];   // 16 KB
  __shared__ __align__(32) _Float16 sW2p[4 * 4 * 32 * 16];   // 16 KB
  __shared__ __align__(32) _Float16 sH1[WAVES][16 * H1S];    // 18 KB
  __shared__ __align__(32) _Float16 sCN[WAVES][16 * CNS];    // 10 KB, cin then h2
  __shared__ int   sI0[WAVES][16][3];
  __shared__ int   sI1[WAVES][16][3];
  __shared__ float sFR[WAVES][16][3];

  const int tid  = threadIdx.x;
  const int wave = tid >> 5;
  const int lane = tid & 31;
  const int l    = lane & 15;
  const int hw   = lane >> 4;   // lane-half (0/1)

  // ---- stage weights f32->f16, swizzled to fragment order (cW1 K padded 35->64) ----
  for (int idx = tid; idx < 2 * 8 * 32 * 16; idx += BLOCK) {
    int i = idx & 15, ln = (idx >> 4) & 31, nt = (idx >> 9) & 7, kc = idx >> 12;
    int K = kc * 32 + (ln >> 4) * 16 + i;
    int N = nt * 16 + (ln & 15);
    sW1p[idx] = (_Float16)((K < 35) ? cW1[K * 128 + N] : 0.0f);
  }
  for (int idx = tid; idx < 4 * 4 * 32 * 16; idx += BLOCK) {
    int i = idx & 15, ln = (idx >> 4) & 31, nt = (idx >> 9) & 3, kc = idx >> 11;
    int K = kc * 32 + (ln >> 4) * 16 + i;
    int N = nt * 16 + (ln & 15);
    sW2p[idx] = (_Float16)cW2[K * 64 + N];
  }
  __syncthreads();

  // ---- phase A: per-sample point, density trilerp, density MLP ----
  const int base = blockIdx.x * SPB + wave * 16;  // 16 samples, all within one ray
  const int ray  = base >> 6;
  const int si   = (base & 63) + l;

  float ox = rays_o[ray*3+0], oy = rays_o[ray*3+1], oz = rays_o[ray*3+2];
  float dx = rays_d[ray*3+0], dy = rays_d[ray*3+1], dz = rays_d[ray*3+2];
  float t  = (float)si * (1.0f / 63.0f);
  float zv = 2.0f * (1.0f - t) + 6.0f * t;
  float px = ox + dx * zv, py = oy + dy * zv, pz = oz + dz * zv;

  float lo0 = aabb[0], lo1 = aabb[1], lo2 = aabb[2];
  float up0 = aabb[3], up1 = aabb[4], up2 = aabb[5];
  float cx = fminf(fmaxf((px - lo0) / (up0 - lo0) * (G - 1), 0.0f), (float)(G - 1));
  float cy = fminf(fmaxf((py - lo1) / (up1 - lo1) * (G - 1), 0.0f), (float)(G - 1));
  float cz = fminf(fmaxf((pz - lo2) / (up2 - lo2) * (G - 1), 0.0f), (float)(G - 1));
  float fxf = floorf(cx), fyf = floorf(cy), fzf = floorf(cz);
  float fx = cx - fxf, fy = cy - fyf, fz = cz - fzf;
  int x0 = (int)fxf, y0 = (int)fyf, z0 = (int)fzf;
  int x1 = (int)ceilf(cx), y1 = (int)ceilf(cy), z1 = (int)ceilf(cz);

  #define GIDX(X,Y,Z) ((((X) * G + (Y)) * G) + (Z))
  float c000 = dgrid[GIDX(x0,y0,z0)], c001 = dgrid[GIDX(x0,y0,z1)];
  float c010 = dgrid[GIDX(x0,y1,z0)], c011 = dgrid[GIDX(x0,y1,z1)];
  float c100 = dgrid[GIDX(x1,y0,z0)], c101 = dgrid[GIDX(x1,y0,z1)];
  float c110 = dgrid[GIDX(x1,y1,z0)], c111 = dgrid[GIDX(x1,y1,z1)];
  float c00 = c000 * (1.f - fx) + c100 * fx;
  float c01 = c001 * (1.f - fx) + c101 * fx;
  float c10 = c010 * (1.f - fx) + c110 * fx;
  float c11 = c011 * (1.f - fx) + c111 * fx;
  float c0 = c00 * (1.f - fy) + c10 * fy;
  float c1 = c01 * (1.f - fy) + c11 * fy;
  float dfeat = c0 * (1.f - fz) + c1 * fz;

  // density MLP (4 -> 64 -> 1): K too small for WMMA, VALU with uniform (scalar) weights
  float spre = db2[0];
  for (int j = 0; j < 64; ++j) {
    float h = px * dW1[j] + py * dW1[64 + j] + pz * dW1[128 + j]
            + dfeat * dW1[192 + j] + db1[j];
    h = fmaxf(h, 0.0f);
    spre += h * dW2[j];
  }
  float sigma = (spre > 20.0f) ? spre : log1pf(expf(spre));

  if (lane < 16) {
    ws_sigma[base + l] = sigma;
    sI0[wave][l][0] = x0; sI0[wave][l][1] = y0; sI0[wave][l][2] = z0;
    sI1[wave][l][0] = x1; sI1[wave][l][1] = y1; sI1[wave][l][2] = z1;
    sFR[wave][l][0] = fx; sFR[wave][l][1] = fy; sFR[wave][l][2] = fz;
    sCN[wave][l * CNS + 0] = (_Float16)px;
    sCN[wave][l * CNS + 1] = (_Float16)py;
    sCN[wave][l * CNS + 2] = (_Float16)pz;
  }
  __syncthreads();

  // ---- phase B: channel-parallel color trilerp (each corner fetch = contiguous 128B) ----
  for (int s = 0; s < 16; ++s) {
    int ax0 = sI0[wave][s][0], ay0 = sI0[wave][s][1], az0 = sI0[wave][s][2];
    int ax1 = sI1[wave][s][0], ay1 = sI1[wave][s][1], az1 = sI1[wave][s][2];
    float gx = sFR[wave][s][0], gy = sFR[wave][s][1], gz = sFR[wave][s][2];
    size_t b000 = (size_t)GIDX(ax0,ay0,az0) * 32, b001 = (size_t)GIDX(ax0,ay0,az1) * 32;
    size_t b010 = (size_t)GIDX(ax0,ay1,az0) * 32, b011 = (size_t)GIDX(ax0,ay1,az1) * 32;
    size_t b100 = (size_t)GIDX(ax1,ay0,az0) * 32, b101 = (size_t)GIDX(ax1,ay0,az1) * 32;
    size_t b110 = (size_t)GIDX(ax1,ay1,az0) * 32, b111 = (size_t)GIDX(ax1,ay1,az1) * 32;
    float v000 = cfeat[b000 + lane], v001 = cfeat[b001 + lane];
    float v010 = cfeat[b010 + lane], v011 = cfeat[b011 + lane];
    float v100 = cfeat[b100 + lane], v101 = cfeat[b101 + lane];
    float v110 = cfeat[b110 + lane], v111 = cfeat[b111 + lane];
    float u00 = v000 * (1.f - gx) + v100 * gx;
    float u01 = v001 * (1.f - gx) + v101 * gx;
    float u10 = v010 * (1.f - gx) + v110 * gx;
    float u11 = v011 * (1.f - gx) + v111 * gx;
    float u0 = u00 * (1.f - gy) + u10 * gy;
    float u1 = u01 * (1.f - gy) + u11 * gy;
    float val = u0 * (1.f - gz) + u1 * gz;
    sCN[wave][s * CNS + 3 + lane] = (_Float16)val;
    if (lane < 29) sCN[wave][s * CNS + 35 + lane] = (_Float16)0.0f;  // K-pad 35..63
  }
  __syncthreads();

  // ---- layer 1: h1 = relu(cin(16x64) @ cW1(64x128) + cb1), WMMA f16->f32 ----
  const _Float16* cinrow = &sCN[wave][l * CNS];       // 160B rows -> 16B-aligned groups
  v16h a0 = cat8(*(const v8h*)(cinrow + hw * 16),
                 *(const v8h*)(cinrow + 8 + hw * 16));
  v16h a1 = cat8(*(const v8h*)(cinrow + 32 + hw * 16),
                 *(const v8h*)(cinrow + 40 + hw * 16));
  #pragma unroll
  for (int nt = 0; nt < 8; ++nt) {
    float bias = cb1[nt * 16 + l];
    v8f c;
    #pragma unroll
    for (int r = 0; r < 8; ++r) c[r] = bias;
    v16h b0 = *(const v16h*)&sW1p[((0 * 8 + nt) * 32 + lane) * 16];
    v16h b1 = *(const v16h*)&sW1p[((1 * 8 + nt) * 32 + lane) * 16];
    c = wmma_f16(a0, b0, c);
    c = wmma_f16(a1, b1, c);
    #pragma unroll
    for (int r = 0; r < 8; ++r) {
      float v = fmaxf(c[r], 0.0f);
      sH1[wave][(hw * 8 + r) * H1S + nt * 16 + l] = (_Float16)v;
    }
  }

  // ---- layer 2: h2 = relu(h1(16x128) @ cW2(128x64) + cb2), WMMA ----
  const _Float16* h1row = &sH1[wave][l * H1S];        // 288B rows -> aligned groups
  v16h A[4];
  #pragma unroll
  for (int k = 0; k < 4; ++k)
    A[k] = cat8(*(const v8h*)(h1row + k * 32 + hw * 16),
                *(const v8h*)(h1row + k * 32 + 8 + hw * 16));
  #pragma unroll
  for (int nt = 0; nt < 4; ++nt) {
    float bias = cb2[nt * 16 + l];
    v8f c;
    #pragma unroll
    for (int r = 0; r < 8; ++r) c[r] = bias;
    #pragma unroll
    for (int k = 0; k < 4; ++k) {
      v16h b = *(const v16h*)&sW2p[((k * 4 + nt) * 32 + lane) * 16];
      c = wmma_f16(A[k], b, c);
    }
    #pragma unroll
    for (int r = 0; r < 8; ++r) {
      float v = fmaxf(c[r], 0.0f);
      sCN[wave][(hw * 8 + r) * CNS + nt * 16 + l] = (_Float16)v;   // h2 reuses cin buffer
    }
  }
  __syncthreads();

  // ---- layer 3: rgb = sigmoid(h2(16x64) @ cW3(64x3) + cb3), VALU (N=3) ----
  if (lane < 16) {
    float r0 = cb3[0], r1 = cb3[1], r2 = cb3[2];
    #pragma unroll
    for (int jc = 0; jc < 8; ++jc) {
      v8h hv = *(const v8h*)&sCN[wave][l * CNS + jc * 8];
      #pragma unroll
      for (int i = 0; i < 8; ++i) {
        float h = (float)hv[i];
        int j = jc * 8 + i;
        r0 += h * cW3[j * 3 + 0];
        r1 += h * cW3[j * 3 + 1];
        r2 += h * cW3[j * 3 + 2];
      }
    }
    int g = base + l;
    ws_rgb[g * 3 + 0] = 1.0f / (1.0f + expf(-r0));
    ws_rgb[g * 3 + 1] = 1.0f / (1.0f + expf(-r1));
    ws_rgb[g * 3 + 2] = 1.0f / (1.0f + expf(-r2));
  }
}

// ---- per-ray volumetric composite ----
__global__ __launch_bounds__(256) void composite_kernel(
    const float* __restrict__ ws_sigma, const float* __restrict__ ws_rgb,
    float* __restrict__ out)
{
  int ray = blockIdx.x * blockDim.x + threadIdx.x;
  if (ray >= NRAYS) return;
  float T = 1.0f, ra = 0.f, ga = 0.f, ba = 0.f, da = 0.f;
  for (int i = 0; i < NSAMP; ++i) {
    float t = (float)i * (1.0f / 63.0f);
    float z = 2.0f + 4.0f * t;
    float dist = (i < NSAMP - 1) ? (4.0f / 63.0f) : 1e10f;
    float sg = ws_sigma[ray * NSAMP + i];
    float alpha = 1.0f - expf(-sg * dist);
    float w = alpha * T;
    ra += w * ws_rgb[(ray * NSAMP + i) * 3 + 0];
    ga += w * ws_rgb[(ray * NSAMP + i) * 3 + 1];
    ba += w * ws_rgb[(ray * NSAMP + i) * 3 + 2];
    da += w * z;
    T *= (1.0f - alpha + 1e-10f);
  }
  out[ray * 3 + 0] = ra;
  out[ray * 3 + 1] = ga;
  out[ray * 3 + 2] = ba;
  out[NRAYS * 3 + ray] = da;
}

extern "C" void kernel_launch(void* const* d_in, const int* in_sizes, int n_in,
                              void* d_out, int out_size, void* d_ws, size_t ws_size,
                              hipStream_t stream) {
  (void)in_sizes; (void)n_in; (void)out_size; (void)ws_size;
  const float* rays_o = (const float*)d_in[0];
  const float* rays_d = (const float*)d_in[1];
  const float* aabb   = (const float*)d_in[2];
  const float* dgrid  = (const float*)d_in[3];
  const float* cfeat  = (const float*)d_in[4];
  const float* dW1    = (const float*)d_in[5];
  const float* db1    = (const float*)d_in[6];
  const float* dW2    = (const float*)d_in[7];
  const float* db2    = (const float*)d_in[8];
  const float* cW1    = (const float*)d_in[9];
  const float* cb1    = (const float*)d_in[10];
  const float* cW2    = (const float*)d_in[11];
  const float* cb2    = (const float*)d_in[12];
  const float* cW3    = (const float*)d_in[13];
  const float* cb3    = (const float*)d_in[14];

  float* ws       = (float*)d_ws;
  float* ws_sigma = ws;                 // TOTAL floats
  float* ws_rgb   = ws + TOTAL;         // TOTAL*3 floats

  nerf_point_kernel<<<TOTAL / SPB, BLOCK, 0, stream>>>(
      rays_o, rays_d, aabb, dgrid, cfeat,
      dW1, db1, dW2, db2, cW1, cb1, cW2, cb2, cW3, cb3,
      ws_sigma, ws_rgb);

  composite_kernel<<<(NRAYS + 255) / 256, 256, 0, stream>>>(
      ws_sigma, ws_rgb, (float*)d_out);
}
